// NeuralNetwork_31275951850011
// MI455X (gfx1250) — compile-verified, matrix operand order checked by reading, not run on hardware
//
#include <hip/hip_runtime.h>

typedef _Float16 half_t;
typedef __attribute__((ext_vector_type(16))) _Float16 v16h;
typedef __attribute__((ext_vector_type(8)))  _Float16 v8h;
typedef __attribute__((ext_vector_type(4)))  _Float16 v4h;
typedef __attribute__((ext_vector_type(8)))  float    v8f;
typedef __attribute__((ext_vector_type(4)))  float    v4f;

#define DWID   64
#define NLAY   8
#define NCONN  28
#define WSTR   72                       // padded LDS row stride in halves (144B = 9*16B, bank-spread)
#define WAVES  8
#define ROWS_PER_BLOCK (WAVES * 16)

// LDS layout (bytes):
//  [0, 258048)          weights  : NCONN * 64 rows * WSTR halves (f16)
//  [258048, 265216)     bias     : NCONN * 64 floats
//  [265216, 283648)     staging  : WAVES * 16 * WSTR halves (f16)
#define LDS_W_BYTES   (NCONN * 64 * WSTR * 2)
#define LDS_B_BYTES   (NCONN * 64 * 4)
#define LDS_S_BYTES   (WAVES * 16 * WSTR * 2)
#define LDS_TOTAL     (LDS_W_BYTES + LDS_B_BYTES + LDS_S_BYTES)

// Load one 16x32 f16 A/B fragment (v16h per lane) from an LDS tile with row
// stride WSTR halves. Per-lane layout for V_WMMA_*_16X16X32_F16:
//   row  = lane & 15
//   VGPR0-3 halves = K [base, base+8),  VGPR4-7 halves = K [base+16, base+24)
//   where base = (lane>>4)*8.  => two contiguous 16B chunks per lane.
__device__ __forceinline__ v16h ld_frag_lds(const half_t* row0, int lane) {
    const half_t* p = row0 + (lane & 15) * WSTR + ((lane >> 4) << 3);
    v8h lo = *(const v8h*)p;
    v8h hi = *(const v8h*)(p + 16);
    return __builtin_shufflevector(lo, hi, 0,1,2,3,4,5,6,7,8,9,10,11,12,13,14,15);
}

// Same fragment shape, but sourced from row-major f32 global memory (the input
// x tile), converting to f16 on the fly. kc selects K chunk 0..31 / 32..63.
__device__ __forceinline__ v16h ld_frag_x(const float* tile, int lane, int kc) {
    const float* p = tile + (size_t)(lane & 15) * DWID + kc * 32 + ((lane >> 4) << 3);
    v4f a = *(const v4f*)(p + 0);
    v4f b = *(const v4f*)(p + 4);
    v4f c = *(const v4f*)(p + 16);
    v4f d = *(const v4f*)(p + 20);
    v16h r;
#pragma unroll
    for (int e = 0; e < 4; ++e) {
        r[e]      = (half_t)a[e];
        r[4 + e]  = (half_t)b[e];
        r[8 + e]  = (half_t)c[e];
        r[12 + e] = (half_t)d[e];
    }
    return r;
}

__global__ __launch_bounds__(256) void dag_mlp_fused(
        const float* __restrict__ x,     // [B, 64]
        const float* __restrict__ Ws,    // [28, 64, 64] (torch Linear: [out,in])
        const float* __restrict__ bs,    // [28, 64]
        float* __restrict__ out,         // [B, 64]
        int B)
{
    extern __shared__ __align__(16) char smem[];
    half_t* wl  = (half_t*)smem;
    float*  bl  = (float*)(smem + LDS_W_BYTES);
    half_t* stg = (half_t*)(smem + LDS_W_BYTES + LDS_B_BYTES);

    const int tid  = threadIdx.x;
    const int lane = tid & 31;
    const int wave = tid >> 5;

    // ---- cooperative: all 28 weight matrices -> f16 LDS (padded rows) ----
    for (int idx = tid; idx < NCONN * 64 * 16; idx += blockDim.x) {
        const int cr = idx >> 4;              // c*64 + n   (row)
        const int k4 = (idx & 15) << 2;       // k chunk of 4
        v4f w = *(const v4f*)(Ws + (size_t)cr * DWID + k4);
        v4h h;
#pragma unroll
        for (int e = 0; e < 4; ++e) h[e] = (half_t)w[e];
        *(v4h*)(wl + cr * WSTR + k4) = h;
    }
    for (int idx = tid; idx < NCONN * 64; idx += blockDim.x) bl[idx] = bs[idx];
    __syncthreads();

    // ---- load this wave's 16-row input tile as A-fragments (layer 0) ----
    const size_t rowbase = ((size_t)blockIdx.x * WAVES + wave) * 16;
    v16h act[NLAY][2];
    {
        const float* xb = x + rowbase * DWID;
        act[0][0] = ld_frag_x(xb, lane, 0);
        act[0][1] = ld_frag_x(xb, lane, 1);
    }

    half_t* mystg = stg + wave * 16 * WSTR;
    const int nsel = lane & 15;   // column-within-N-tile held by this lane (C layout)
    const int mhi  = (lane >> 4) << 3;

    // ---- fused DAG: build layers 1..7 ----
#pragma unroll
    for (int j = 1; j < NLAY; ++j) {
        v8f sum[4];
#pragma unroll
        for (int nt = 0; nt < 4; ++nt)
#pragma unroll
            for (int e = 0; e < 8; ++e) sum[nt][e] = 0.0f;

#pragma unroll
        for (int i = 0; i < j; ++i) {
            const int c = i * 7 - (i * (i - 1)) / 2 + (j - i - 1);
            const half_t* wc = wl + c * (64 * WSTR);
            const float*  bc = bl + c * 64;
#pragma unroll
            for (int nt = 0; nt < 4; ++nt) {
                const float bval = bc[nt * 16 + nsel];
                v8f acc;
#pragma unroll
                for (int e = 0; e < 8; ++e) acc[e] = bval;     // C starts at bias
                v16h b0 = ld_frag_lds(wc + nt * 16 * WSTR,      lane);
                v16h b1 = ld_frag_lds(wc + nt * 16 * WSTR + 32, lane);
                acc = __builtin_amdgcn_wmma_f32_16x16x32_f16(
                        false, act[i][0], false, b0, (short)0, acc, false, false);
                acc = __builtin_amdgcn_wmma_f32_16x16x32_f16(
                        false, act[i][1], false, b1, (short)0, acc, false, false);
#pragma unroll
                for (int e = 0; e < 8; ++e) sum[nt][e] += fmaxf(acc[e], 0.0f);
            }
        }

        if (j < NLAY - 1) {
            // round-trip C-layout f32 -> f16 staging -> A-fragment registers
            __syncthreads();   // previous layer's staging reads are done
#pragma unroll
            for (int nt = 0; nt < 4; ++nt)
#pragma unroll
                for (int e = 0; e < 8; ++e) {
                    const int m = e + mhi;
                    const int n = nt * 16 + nsel;
                    mystg[m * WSTR + n] = (half_t)sum[nt][e];
                }
            __syncthreads();   // stores visible before fragment reads
            act[j][0] = ld_frag_lds(mystg,      lane);
            act[j][1] = ld_frag_lds(mystg + 32, lane);
        } else {
            // final layer: f32 straight to global
            float* ob = out + rowbase * DWID;
#pragma unroll
            for (int nt = 0; nt < 4; ++nt)
#pragma unroll
                for (int e = 0; e < 8; ++e)
                    ob[(size_t)(e + mhi) * DWID + nt * 16 + nsel] = sum[nt][e];
        }
    }
}

extern "C" void kernel_launch(void* const* d_in, const int* in_sizes, int n_in,
                              void* d_out, int out_size, void* d_ws, size_t ws_size,
                              hipStream_t stream) {
    const float* x  = (const float*)d_in[0];
    const float* Ws = (const float*)d_in[1];
    const float* bs = (const float*)d_in[2];
    float* out = (float*)d_out;

    const int B = in_sizes[0] / DWID;                 // 262144
    const int grid = (B + ROWS_PER_BLOCK - 1) / ROWS_PER_BLOCK;   // 2048

    dag_mlp_fused<<<grid, 256, LDS_TOTAL, stream>>>(x, Ws, bs, out, B);
}
